// NaiveAttention_14826227105870
// MI455X (gfx1250) — matrix operand order repaired and checked by reading, not compiled, not run on hardware
//
#include <hip/hip_runtime.h>

typedef __attribute__((ext_vector_type(16))) __bf16 v16bf;
typedef __attribute__((ext_vector_type(8)))  __bf16 v8bf;
typedef __attribute__((ext_vector_type(4)))  __bf16 v4bf;
typedef __attribute__((ext_vector_type(8)))  float  v8f;
typedef __attribute__((ext_vector_type(4)))  float  v4f;

#define BB 8
#define CC 64
#define AA 32
#define NN 4096
#define NTILES 256            // NN/16
#define SC_STRIDE 4100        // 4096 + 4-word pad (bank-conflict free, 16B aligned)
#define LDS_BYTES (16*SC_STRIDE*4 + 8*32*8*4)   // 262400 + 8192 = 270592 B

// wave32 reductions via ds_swizzle (SWAPX16/8/4/2/1)
__device__ __forceinline__ float wave_max32(float v) {
  v = fmaxf(v, __int_as_float(__builtin_amdgcn_ds_swizzle(__float_as_int(v), 0x401F)));
  v = fmaxf(v, __int_as_float(__builtin_amdgcn_ds_swizzle(__float_as_int(v), 0x201F)));
  v = fmaxf(v, __int_as_float(__builtin_amdgcn_ds_swizzle(__float_as_int(v), 0x101F)));
  v = fmaxf(v, __int_as_float(__builtin_amdgcn_ds_swizzle(__float_as_int(v), 0x081F)));
  v = fmaxf(v, __int_as_float(__builtin_amdgcn_ds_swizzle(__float_as_int(v), 0x041F)));
  return v;
}
__device__ __forceinline__ float wave_sum32(float v) {
  v += __int_as_float(__builtin_amdgcn_ds_swizzle(__float_as_int(v), 0x401F));
  v += __int_as_float(__builtin_amdgcn_ds_swizzle(__float_as_int(v), 0x201F));
  v += __int_as_float(__builtin_amdgcn_ds_swizzle(__float_as_int(v), 0x101F));
  v += __int_as_float(__builtin_amdgcn_ds_swizzle(__float_as_int(v), 0x081F));
  v += __int_as_float(__builtin_amdgcn_ds_swizzle(__float_as_int(v), 0x041F));
  return v;
}

// ---------------------------------------------------------------------------
// Kernel 1: 1x1-conv projections g = Wg x + bg, h = Wh x + bh, plus bf16 x,
// all stored pre-swizzled into CDNA5 WMMA fragment layouts (ISA 7.12.2):
//   asw : A-fragment of g^T tiles [16 n x 32 a]   -> [b][nt][lane][16 halves]
//   hsw : B-fragment of h  tiles  [32 a x 16 m]   -> [b][mt][lane][16 halves]
//   xsw : A-fragment of x  tiles  [16 c x 32 m]   -> [b][ct][mc][lane][16 halves]
// ---------------------------------------------------------------------------
__global__ __launch_bounds__(256) void proj_kernel(
    const float* __restrict__ x,  const float* __restrict__ Wg,
    const float* __restrict__ bg, const float* __restrict__ Wh,
    const float* __restrict__ bh,
    __bf16* __restrict__ asw, __bf16* __restrict__ hsw, __bf16* __restrict__ xsw)
{
  __shared__ float sWg[AA*CC], sWh[AA*CC], sbg[AA], sbh[AA];
  const int t = threadIdx.x;
  for (int idx = t; idx < AA*CC; idx += 256) { sWg[idx] = Wg[idx]; sWh[idx] = Wh[idx]; }
  if (t < AA) { sbg[t] = bg[t]; sbh[t] = bh[t]; }
  __syncthreads();

  const int gid = blockIdx.x * 256 + t;
  const int b = gid >> 12, n = gid & (NN - 1);

  float xv[CC];
  const float* xp = x + ((size_t)b * CC) * NN + n;
  #pragma unroll
  for (int c = 0; c < CC; ++c) xv[c] = xp[(size_t)c * NN];

  // xsw: A-fragment (16-bit A 16x32 layout). Row = c%16, K = m%32.
  {
    const int mc = n >> 5, k = n & 31;
    const int Ladd = (k < 16) ? 0 : 16;
    const int ii = k & 15;
    #pragma unroll
    for (int c = 0; c < CC; ++c) {
      const int ct = c >> 4;
      const int L  = (c & 15) + Ladd;
      xsw[((((size_t)(b * 4 + ct)) * 128 + mc) * 32 + L) * 16 + ii] = (__bf16)xv[c];
    }
  }

  const int nt = n >> 4, r = n & 15;
  for (int a = 0; a < AA; ++a) {
    float ga = sbg[a], ha = sbh[a];
    #pragma unroll
    for (int c = 0; c < CC; ++c) { ga += sWg[a*CC + c] * xv[c]; ha += sWh[a*CC + c] * xv[c]; }
    // asw: A-fragment, row = n%16, K = a
    const int La = r + (((a & 15) < 8) ? 0 : 16);
    const int ia = (a < 16) ? (a & 7) : ((a & 7) + 8);
    asw[(((size_t)(b * NTILES + nt)) * 32 + La) * 16 + ia] = (__bf16)ga;
    // hsw: B-fragment, K = a, col = m%16
    const int Lh = r + (((a & 15) < 8) ? 0 : 16);
    hsw[(((size_t)(b * NTILES + nt)) * 32 + Lh) * 16 + ia] = (__bf16)ha;
  }
}

// ---------------------------------------------------------------------------
// Kernel 2: fused scores -> softmax(alpha out) -> r = alpha @ x^T -> +x, ReLU
// One block per (b, 16-row n-tile). 8 wave32s. 270KB dynamic LDS.
// ---------------------------------------------------------------------------
__global__ __launch_bounds__(256) void attn_kernel(
    const float* __restrict__ x,
    const __bf16* __restrict__ asw, const __bf16* __restrict__ hsw,
    const __bf16* __restrict__ xsw,
    float* __restrict__ outR, float* __restrict__ outAlpha)
{
  extern __shared__ char smem[];
  float* sc   = (float*)smem;                       // [16][SC_STRIDE] scores/alpha
  float* pbuf = (float*)(smem + 16*SC_STRIDE*4);    // [4 waves][32 lanes][8] partials

  const int t = threadIdx.x;
  const int wave = t >> 5, lane = t & 31;
  const int b = blockIdx.x >> 8, nt = blockIdx.x & 255;
  const int n0 = nt * 16;
  const int lr = lane & 15;
  const bool hi = lane >= 16;

  // ---------------- Phase A: scores tile [16 x 4096] ----------------
  // A-fragment (g^T tile) is shared across the whole strip.
  const v16bf afrag = *(const v16bf*)(asw + (((size_t)(b * NTILES + nt)) * 32 + lane) * 16);
  const __bf16* hp = hsw + (((size_t)(b * NTILES + wave * 32)) * 32 + lane) * 16;
  const int rowbase = hi ? 8 : 0;
  #pragma unroll 2
  for (int mt = 0; mt < 32; ++mt) {
    const v16bf bfrag = *(const v16bf*)hp;
    hp += 32 * 16;
    __builtin_prefetch((const void*)hp, 0, 3);
    v8f acc = {};
    acc = __builtin_amdgcn_wmma_f32_16x16x32_bf16(false, afrag, false, bfrag,
                                                  (short)0, acc, false, false);
    const int col = (wave * 32 + mt) * 16 + lr;
    #pragma unroll
    for (int v = 0; v < 8; ++v) sc[(rowbase + v) * SC_STRIDE + col] = acc[v];
  }
  __syncthreads();

  // ---------------- Phase B: softmax, 2 rows per wave, 16B vectors ----------------
  // Last pass also compacts the row to bf16 in-place (B-fragment order for phase C);
  // bf16 writes (8B/lane/iter) strictly trail the f32 reads (16B/lane/iter).
  #pragma unroll
  for (int rr = 0; rr < 2; ++rr) {
    const int row = wave * 2 + rr;
    float* rp = sc + row * SC_STRIDE;
    v4f*  rp4 = (v4f*)rp;                           // 1024 v4f per row, 16B aligned
    v4bf* rb4 = (v4bf*)rp;                          // bf16 alias, first half of row
    float m = -3.402823466e38f;
    #pragma unroll 4
    for (int it = 0; it < 32; ++it) {
      const v4f q = rp4[lane + 32 * it];
      m = fmaxf(m, fmaxf(fmaxf(q.x, q.y), fmaxf(q.z, q.w)));
    }
    m = wave_max32(m);
    float s = 0.f;
    #pragma unroll 2
    for (int it = 0; it < 32; ++it) {
      v4f q = rp4[lane + 32 * it];
      q.x = __expf(q.x - m); q.y = __expf(q.y - m);
      q.z = __expf(q.z - m); q.w = __expf(q.w - m);
      s += (q.x + q.y) + (q.z + q.w);
      rp4[lane + 32 * it] = q;
    }
    s = wave_sum32(s);
    const float inv = 1.0f / s;
    v4f* ga4 = (v4f*)(outAlpha + ((size_t)(b * NN + n0 + row)) * NN);
    for (int it = 0; it < 32; ++it) {
      const int j = lane + 32 * it;
      v4f q = rp4[j];
      q.x *= inv; q.y *= inv; q.z *= inv; q.w *= inv;
      __builtin_nontemporal_store(q, ga4 + j);      // 537MB stream: keep out of L2
      const v4bf qb = { (__bf16)q.x, (__bf16)q.y, (__bf16)q.z, (__bf16)q.w };
      rb4[j] = qb;                                  // in-place f32 -> bf16 compaction
      asm volatile("" ::: "memory");                // pin read/write order (no insns)
    }
  }
  __syncthreads();

  // ---------------- Phase C: r[c,n] = sum_m x[c,m] * alpha[n,m] ----------------
  // x is the A-fragment (pre-swizzled global); alpha B-fragment is 32 contiguous
  // bytes of the lane's bf16 LDS row -> 2x ds_load_b128, no conversion VALU.
  // Two independent accumulators (even/odd m-chunks) keep WMMAs dependency-free.
  const int ct = wave & 3;
  const int mcBeg = (wave >> 2) * 64;
  v8f acc0 = {}, acc1 = {};
  const __bf16* xpf = xsw + ((((size_t)(b * 4 + ct)) * 128 + mcBeg) * 32 + lane) * 16;
  const __bf16* brow = (const __bf16*)(sc + lr * SC_STRIDE);
  const int boff = hi ? 16 : 0;
  #pragma unroll 2
  for (int u = 0; u < 32; ++u) {
    const int mc = mcBeg + 2 * u;
    const v16bf af0 = *(const v16bf*)xpf;
    const v16bf af1 = *(const v16bf*)(xpf + 32 * 16);
    xpf += 2 * 32 * 16;
    __builtin_prefetch((const void*)xpf, 0, 3);
    const v8bf b0lo = *(const v8bf*)(brow + mc * 32 + boff);
    const v8bf b0hi = *(const v8bf*)(brow + mc * 32 + boff + 8);
    const v8bf b1lo = *(const v8bf*)(brow + (mc + 1) * 32 + boff);
    const v8bf b1hi = *(const v8bf*)(brow + (mc + 1) * 32 + boff + 8);
    const v16bf bf0 = __builtin_shufflevector(b0lo, b0hi,
        0,1,2,3,4,5,6,7,8,9,10,11,12,13,14,15);
    const v16bf bf1 = __builtin_shufflevector(b1lo, b1hi,
        0,1,2,3,4,5,6,7,8,9,10,11,12,13,14,15);
    acc0 = __builtin_amdgcn_wmma_f32_16x16x32_bf16(false, af0, false, bf0,
                                                   (short)0, acc0, false, false);
    acc1 = __builtin_amdgcn_wmma_f32_16x16x32_bf16(false, af1, false, bf1,
                                                   (short)0, acc1, false, false);
  }
  v8f acc;
  #pragma unroll
  for (int v = 0; v < 8; ++v) acc[v] = acc0[v] + acc1[v];

  // combine the two m-halves through LDS, then residual + ReLU + NT store
  if (wave >= 4) {
    float* pp = pbuf + (((wave - 4) * 32 + lane)) * 8;
    #pragma unroll
    for (int v = 0; v < 8; ++v) pp[v] = acc[v];
  }
  __syncthreads();
  if (wave < 4) {
    const float* pp = pbuf + ((wave * 32 + lane)) * 8;
    #pragma unroll
    for (int v = 0; v < 8; ++v) {
      const float rv = acc[v] + pp[v];
      const int c = ct * 16 + (hi ? 8 : 0) + v;
      const int n = n0 + lr;
      const size_t oi = ((size_t)(b * CC + c)) * NN + n;
      const float o = rv + x[oi];
      __builtin_nontemporal_store(fmaxf(o, 0.0f), outR + oi);
    }
  }
}

// ---------------------------------------------------------------------------
extern "C" void kernel_launch(void* const* d_in, const int* in_sizes, int n_in,
                              void* d_out, int out_size, void* d_ws, size_t ws_size,
                              hipStream_t stream) {
  (void)in_sizes; (void)n_in; (void)out_size; (void)ws_size;
  const float* x  = (const float*)d_in[0];
  const float* Wg = (const float*)d_in[1];
  const float* bg = (const float*)d_in[2];
  const float* Wh = (const float*)d_in[3];
  const float* bh = (const float*)d_in[4];

  __bf16* asw = (__bf16*)d_ws;                              // 1,048,576 elems (2MB)
  __bf16* hsw = asw + (size_t)BB * NTILES * 32 * 16;        // 2MB
  __bf16* xsw = hsw + (size_t)BB * NTILES * 32 * 16;        // 4MB

  float* outR = (float*)d_out;                              // [B,C,H,W]
  float* outA = outR + (size_t)BB * CC * NN;                // [B,N,N]

  proj_kernel<<<(BB * NN) / 256, 256, 0, stream>>>(x, Wg, bg, Wh, bh, asw, hsw, xsw);
  attn_kernel<<<BB * NTILES, 256, LDS_BYTES, stream>>>(x, asw, hsw, xsw, outR, outA);
}
